// TGNRiskPredictor_52544629899847
// MI455X (gfx1250) — matrix operand order repaired and checked by reading, not compiled, not run on hardware
//
#include <hip/hip_runtime.h>
#include <hip/hip_bf16.h>

typedef __attribute__((ext_vector_type(16))) __bf16 v16bf;
typedef __attribute__((ext_vector_type(8)))  __bf16 v8bf;
typedef __attribute__((ext_vector_type(8)))  float  v8f;

#define PW 832   // projection atlas width: q(256) | k(256) | v(256) | xr(64)

static __device__ __forceinline__ __bf16 f2bf(float f) {
    unsigned u = __float_as_uint(f);
    unsigned r = (u + 0x7FFFu + ((u >> 16) & 1u)) >> 16;   // round-to-nearest-even
    unsigned short s = (unsigned short)r;
    __bf16 b;
    __builtin_memcpy(&b, &s, sizeof(b));
    return b;
}

// map float -> unsigned with order preserved, so atomicMax(u32) == float max
static __device__ __forceinline__ unsigned enc_ord(float f) {
    unsigned u = __float_as_uint(f);
    return (u & 0x80000000u) ? ~u : (u | 0x80000000u);
}
static __device__ __forceinline__ float dec_ord(unsigned e) {
    unsigned u = (e & 0x80000000u) ? (e & 0x7FFFFFFFu) : ~e;
    return __uint_as_float(u);
}
#define ENC_NEG_INF 0x007FFFFFu   // enc_ord(-inf)

// ---------------- 1) embedding gather directly to bf16: hbf = bf16(emb[x]) ---------
__global__ void gather_bf16_kernel(const int* __restrict__ x, const float* __restrict__ emb,
                                   __bf16* __restrict__ hbf, int n) {
    int i = blockIdx.x * blockDim.x + threadIdx.x;   // over n*8 (8 elems each)
    if (i >= n * 8) return;
    int node = i >> 3, seg = i & 7;
    const float* ep = emb + (size_t)x[node] * 64 + seg * 8;
    v8bf o;
    #pragma unroll
    for (int j = 0; j < 8; ++j) o[j] = f2bf(ep[j]);
    *(v8bf*)(hbf + (size_t)node * 64 + seg * 8) = o;
}

// ---------------- 2) transposed bf16 weight atlas Wt[832][64] + bias atlas ---------
// rows 0-255 = Wq cols, 256-511 = Wk cols, 512-767 = Wv cols, 768-831 = Wskip cols.
__global__ void wt_bf16_kernel(const float* __restrict__ Wq, const float* __restrict__ Wk,
                               const float* __restrict__ Wv, const float* __restrict__ Ws,
                               const float* __restrict__ bq, const float* __restrict__ bk,
                               const float* __restrict__ bv, const float* __restrict__ bs,
                               __bf16* __restrict__ Wt, float* __restrict__ bcat) {
    int i = blockIdx.x * blockDim.x + threadIdx.x;   // over 832*64
    if (i >= PW * 64) return;
    int r = i >> 6, kd = i & 63;
    float val, bval;
    if (r < 256)      { val = Wq[kd * 256 + r];         bval = bq[r]; }
    else if (r < 512) { val = Wk[kd * 256 + (r - 256)]; bval = bk[r - 256]; }
    else if (r < 768) { val = Wv[kd * 256 + (r - 512)]; bval = bv[r - 512]; }
    else              { val = Ws[kd * 64  + (r - 768)]; bval = bs[r - 768]; }
    Wt[i] = f2bf(val);
    if (kd == 0) bcat[r] = bval;
}

__global__ void fill_kernel(unsigned* __restrict__ p, unsigned val, int n) {
    int i = blockIdx.x * blockDim.x + threadIdx.x;
    if (i < n) p[i] = val;
}

// ---------------- 3) fused projections via WMMA ------------------------------------
// One wave per 16-row node tile x 4 output tiles (grid.y = 13 -> 52 tiles total).
// A fragment (16x64 bf16) loaded once as 4x b128 and reused for all 4 tiles; each
// B fragment is 4x b128 from the transposed atlas. Two chained 16x16x32 bf16 WMMAs.
// Branch-free epilogue on the full-tile (uniform) fast path: 8 stores at immediate
// offsets r*PW*4 from one base address.
__global__ void proj_gemm_kernel(const __bf16* __restrict__ hbf, const __bf16* __restrict__ Wt,
                                 const float* __restrict__ bcat,
                                 float* __restrict__ proj, int n) {
    const int lane = threadIdx.x & 31;
    const int row0 = blockIdx.x * 16;
    const int hi   = lane >> 4;                 // 0: lanes 0-15, 1: lanes 16-31
    const bool full = (row0 + 16 <= n);         // wave-uniform tail predicate

    int arow = row0 + (lane & 15); if (arow >= n) arow = n - 1;
    // ISA 7.12.2 A layout: lane group hi uses K chunks {hi*8+0..7, hi*8+16..23} (+32 for kk=1)
    const __bf16* ap = hbf + (size_t)arow * 64 + hi * 8;
    const v8bf a00 = *(const v8bf*)(ap);        // kk=0, VGPR j=0..3
    const v8bf a01 = *(const v8bf*)(ap + 16);   // kk=0, VGPR j=4..7
    const v8bf a10 = *(const v8bf*)(ap + 32);   // kk=1, VGPR j=0..3
    const v8bf a11 = *(const v8bf*)(ap + 48);   // kk=1, VGPR j=4..7
    const v16bf A0 = __builtin_shufflevector(a00, a01, 0,1,2,3,4,5,6,7,8,9,10,11,12,13,14,15);
    const v16bf A1 = __builtin_shufflevector(a10, a11, 0,1,2,3,4,5,6,7,8,9,10,11,12,13,14,15);

    #pragma unroll
    for (int t = 0; t < 4; ++t) {
        const int otile = blockIdx.y * 4 + t;
        const int gcol  = otile * 16 + (lane & 15);      // atlas column (0..831)
        // B layout: lane = column; lane group hi holds K {hi*8+0..7, hi*8+16..23} (+32 for kk=1)
        const __bf16* bp = Wt + (size_t)gcol * 64 + hi * 8;
        const v8bf b00 = *(const v8bf*)(bp);
        const v8bf b01 = *(const v8bf*)(bp + 16);
        const v8bf b10 = *(const v8bf*)(bp + 32);
        const v8bf b11 = *(const v8bf*)(bp + 48);
        const v16bf B0 = __builtin_shufflevector(b00, b01, 0,1,2,3,4,5,6,7,8,9,10,11,12,13,14,15);
        const v16bf B1 = __builtin_shufflevector(b10, b11, 0,1,2,3,4,5,6,7,8,9,10,11,12,13,14,15);

        v8f c = {};
        c = __builtin_amdgcn_wmma_f32_16x16x32_bf16(false, A0, false, B0, (short)0, c, false, false);
        c = __builtin_amdgcn_wmma_f32_16x16x32_bf16(false, A1, false, B1, (short)0, c, false, false);

        const float bb = bcat[gcol];
        // C/D layout: VGPR r -> row row0+r (lanes 0-15) / row0+r+8 (lanes 16-31)
        float* outp = proj + (size_t)(row0 + hi * 8) * PW + gcol;
        if (full) {
            #pragma unroll
            for (int r = 0; r < 8; ++r) outp[(size_t)r * PW] = c[r] + bb;   // imm offsets
        } else {
            #pragma unroll
            for (int r = 0; r < 8; ++r)
                if (row0 + r + hi * 8 < n) outp[(size_t)r * PW] = c[r] + bb;
        }
    }
}

// ---------------- 4) edge attention: alpha + segment max ---------------------------
// 16 lanes per (edge, head): coalesced float4 loads, shuffle-tree reduction.
// proj atlas: q at col 0, k at col 256.
__global__ void attn_alpha_kernel(const int* __restrict__ src, const int* __restrict__ dst,
                                  const float* __restrict__ proj,
                                  float* __restrict__ alpha, unsigned* __restrict__ m_enc,
                                  int e_total) {
    int idx = blockIdx.x * blockDim.x + threadIdx.x;   // over E*H*16
    if (idx >= e_total * 4 * 16) return;
    const int pair = idx >> 4, sub = idx & 15;
    const int e = pair >> 2, hd = pair & 3;
    const int s = src[e], d = dst[e];
    const float4 a = *(const float4*)(proj + (size_t)d * PW + hd * 64 + sub * 4);        // q[d]
    const float4 b = *(const float4*)(proj + (size_t)s * PW + 256 + hd * 64 + sub * 4);  // k[s]
    float acc = a.x * b.x + a.y * b.y + a.z * b.z + a.w * b.w;
    #pragma unroll
    for (int off = 8; off > 0; off >>= 1) acc += __shfl_xor(acc, off, 32);
    if (sub == 0) {
        acc *= 0.125f;                  // 1/sqrt(64)
        alpha[pair] = acc;
        atomicMax(&m_enc[d * 4 + hd], enc_ord(acc));
    }
}

// ---------------- 5) exp(alpha - max) + segment sum --------------------------------
__global__ void attn_exp_kernel(const int* __restrict__ dst, float* __restrict__ alpha,
                                const unsigned* __restrict__ m_enc,
                                float* __restrict__ denom, int e_total) {
    int idx = blockIdx.x * blockDim.x + threadIdx.x;
    if (idx >= e_total * 4) return;
    const int e = idx >> 2, hd = idx & 3;
    const int d = dst[e];
    const float ev = __expf(alpha[idx] - dec_ord(m_enc[d * 4 + hd]));
    alpha[idx] = ev;                                   // reuse buffer for e
    atomicAdd(&denom[d * 4 + hd], ev);
}

// ---------------- 6) weighted message aggregation ----------------------------------
// 32 lanes per (edge, head): coalesced v reads (atlas col 512), coalesced f32 atomics.
__global__ void attn_agg_kernel(const int* __restrict__ src, const int* __restrict__ dst,
                                const float* __restrict__ proj, const float* __restrict__ alpha,
                                const float* __restrict__ denom, float* __restrict__ outagg,
                                int e_total) {
    int idx = blockIdx.x * blockDim.x + threadIdx.x;   // over E*H*32
    if (idx >= e_total * 4 * 32) return;
    const int pair = idx >> 5, lane = idx & 31;
    const int e = pair >> 2, hd = pair & 3;
    const int s = src[e], d = dst[e];
    const float* vp = proj + (size_t)s * PW + 512 + hd * 64;   // v[s]
    __builtin_prefetch(vp + lane, 0, 0);               // global_prefetch_b8
    const float w = alpha[pair] / denom[d * 4 + hd];
    float* op = outagg + (size_t)d * 256 + hd * 64;
    atomicAdd(&op[lane],      vp[lane]      * w);
    atomicAdd(&op[lane + 32], vp[lane + 32] * w);
}

// ---------------- 7) head mean + beta gate + residual (one wave per node) ----------
__global__ void node_update_kernel(const float* __restrict__ outagg,
                                   const float* __restrict__ proj,
                                   const float* __restrict__ Wbeta,
                                   float* __restrict__ hprime, int n) {
    const int wave = (blockIdx.x * blockDim.x + threadIdx.x) >> 5;
    const int lane = threadIdx.x & 31;
    if (wave >= n) return;
    const float* oa = outagg + (size_t)wave * 256;
    const float* xp = proj + (size_t)wave * PW + 768;          // x_r
    const float o0 = 0.25f * (oa[lane]      + oa[64 + lane] + oa[128 + lane] + oa[192 + lane]);
    const float o1 = 0.25f * (oa[32 + lane] + oa[96 + lane] + oa[160 + lane] + oa[224 + lane]);
    const float x0 = xp[lane];
    const float x1 = xp[lane + 32];
    float s = o0 * Wbeta[lane]              + o1 * Wbeta[lane + 32]
            + x0 * Wbeta[64 + lane]         + x1 * Wbeta[96 + lane]
            + (o0 - x0) * Wbeta[128 + lane] + (o1 - x1) * Wbeta[160 + lane];
    #pragma unroll
    for (int off = 16; off > 0; off >>= 1) s += __shfl_xor(s, off, 32);
    const float beta = 1.f / (1.f + __expf(-s));
    hprime[wave * 64 + lane]      = beta * x0 + (1.f - beta) * o0;
    hprime[wave * 64 + lane + 32] = beta * x1 + (1.f - beta) * o1;
}

// ---------------- 8) task MLP: one wave per task, lane = hidden unit ---------------
__global__ void mlp_kernel(const int* __restrict__ task_idx, const float* __restrict__ hprime,
                           const float* __restrict__ W1, const float* __restrict__ b1,
                           const float* __restrict__ W2, const float* __restrict__ b2,
                           float* __restrict__ out, int t_total) {
    const int wave = (blockIdx.x * blockDim.x + threadIdx.x) >> 5;
    const int lane = threadIdx.x & 31;
    if (wave >= t_total) return;
    const float* hs = hprime + (size_t)task_idx[wave] * 64;
    __builtin_prefetch(hs, 0, 0);                      // global_prefetch_b8
    float acc = b1[lane];
    #pragma unroll 8
    for (int d2 = 0; d2 < 64; ++d2) acc += hs[d2] * W1[d2 * 32 + lane];
    acc = fmaxf(acc, 0.f);
    float r = acc * W2[lane];
    #pragma unroll
    for (int off = 16; off > 0; off >>= 1) r += __shfl_xor(r, off, 32);
    if (lane == 0) out[wave] = 1.f / (1.f + __expf(-(r + b2[0])));
}

extern "C" void kernel_launch(void* const* d_in, const int* in_sizes, int n_in,
                              void* d_out, int out_size, void* d_ws, size_t ws_size,
                              hipStream_t stream) {
    const int*   x     = (const int*)d_in[0];
    const int*   eidx  = (const int*)d_in[1];
    const int*   tidx  = (const int*)d_in[2];
    const float* emb   = (const float*)d_in[3];
    const float* Wq    = (const float*)d_in[4];
    const float* bq    = (const float*)d_in[5];
    const float* Wk    = (const float*)d_in[6];
    const float* bk    = (const float*)d_in[7];
    const float* Wv    = (const float*)d_in[8];
    const float* bv    = (const float*)d_in[9];
    const float* Wskip = (const float*)d_in[10];
    const float* bskip = (const float*)d_in[11];
    const float* Wbeta = (const float*)d_in[12];
    const float* W1    = (const float*)d_in[13];
    const float* b1    = (const float*)d_in[14];
    const float* W2    = (const float*)d_in[15];
    const float* b2    = (const float*)d_in[16];

    const int N = in_sizes[0];          // 50000
    const int E = in_sizes[1] / 2;      // 200000
    const int T = in_sizes[2];          // 10000
    const int* src = eidx;              // edge_index[0]
    const int* dst = eidx + E;          // edge_index[1]

    // workspace layout
    char* ws = (char*)d_ws;
    size_t off = 0;
    auto alloc = [&](size_t bytes) {
        void* p = ws + off;
        off = (off + bytes + 255) & ~(size_t)255;
        return p;
    };
    __bf16*   hbf    = (__bf16*)alloc((size_t)N * 64 * 2);
    __bf16*   Wt     = (__bf16*)alloc((size_t)PW * 64 * 2);
    float*    bcat   = (float*)alloc((size_t)PW * 4);
    float*    proj   = (float*)alloc((size_t)N * PW * 4);     // q | k | v | xr
    float*    alpha  = (float*)alloc((size_t)E * 4 * 4);
    unsigned* m_enc  = (unsigned*)alloc((size_t)N * 4 * 4);
    float*    denom  = (float*)alloc((size_t)N * 4 * 4);
    float*    outagg = (float*)alloc((size_t)N * 256 * 4);
    float*    hprime = (float*)alloc((size_t)N * 64 * 4);
    (void)ws_size; (void)n_in; (void)out_size;

    const int B = 256;
    // 1) gather + weight/bias prep
    gather_bf16_kernel<<<(N * 8 + B - 1) / B, B, 0, stream>>>(x, emb, hbf, N);
    wt_bf16_kernel<<<(PW * 64 + B - 1) / B, B, 0, stream>>>(Wq, Wk, Wv, Wskip,
                                                            bq, bk, bv, bskip, Wt, bcat);
    // init segment buffers
    fill_kernel<<<(N * 4 + B - 1) / B, B, 0, stream>>>(m_enc, ENC_NEG_INF, N * 4);
    fill_kernel<<<(N * 4 + B - 1) / B, B, 0, stream>>>((unsigned*)denom, 0u, N * 4);
    fill_kernel<<<(N * 256 + B - 1) / B, B, 0, stream>>>((unsigned*)outagg, 0u, N * 256);
    // 2) projections (WMMA)
    dim3 ggrid((N + 15) / 16, 13);
    proj_gemm_kernel<<<ggrid, 32, 0, stream>>>(hbf, Wt, bcat, proj, N);
    // 3-5) edge attention
    const int EH = E * 4;
    attn_alpha_kernel<<<((size_t)EH * 16 + B - 1) / B, B, 0, stream>>>(src, dst, proj, alpha, m_enc, E);
    attn_exp_kernel<<<(EH + B - 1) / B, B, 0, stream>>>(dst, alpha, m_enc, denom, E);
    attn_agg_kernel<<<((size_t)EH * 32 + B - 1) / B, B, 0, stream>>>(src, dst, proj, alpha, denom, outagg, E);
    // 6) node update: 8 waves per block
    node_update_kernel<<<(N + 7) / 8, B, 0, stream>>>(outagg, proj, Wbeta, hprime, N);
    // 7) MLP
    mlp_kernel<<<(T + 7) / 8, B, 0, stream>>>(tidx, hprime, W1, b1, W2, b2, (float*)d_out, T);
}
